// DotProductAttention_18133351924414
// MI455X (gfx1250) — matrix operand order repaired and checked by reading, not compiled, not run on hardware
//
#include <hip/hip_runtime.h>

// ---------------------------------------------------------------------------
// Flash attention for MI455X (gfx1250, wave32, WMMA), transposed-score form.
// B=16, S=2048, DK=DV=128, fp32 in/out, bf16 WMMA math with fp32 accumulate.
//
//   S^T = K * Q^T   (A = K rows, B = Q^T columns -> both layouts natural)
//   P^T(C-layout) == P(A-layout) up to register packing -> PV needs no LDS.
//   K/V tiles double-buffered in LDS; staged with global_load_async_to_lds_b128
//   (ASYNCcnt) from a bf16 copy pre-converted into the workspace, overlapped
//   with WMMA compute; s_wait_asynccnt only after the tile's 32 WMMAs.
// ---------------------------------------------------------------------------

typedef __attribute__((ext_vector_type(16))) __bf16 v16bf;
typedef __attribute__((ext_vector_type(8)))  __bf16 v8bf;
typedef __attribute__((ext_vector_type(8)))  float  v8f;
typedef __attribute__((ext_vector_type(4)))  float  v4f;

#define B_    16
#define S_    2048
#define D_    128
#define BM    128      // query rows per workgroup (8 waves x 16 rows)
#define BN    64       // keys per k-tile
#define NEG_BIG (-1.0e9f)
#define LOG2E 1.4426950408889634f

__device__ __forceinline__ __bf16 f2bf(float f) { return (__bf16)f; }

// Async memory->LDS copy, 16B per lane. GVS mode: SGPR 64-bit base + VGPR
// 32-bit byte offset. Tracked by ASYNCcnt.
__device__ __forceinline__ void async_b128(unsigned lds_off, const void* sbase,
                                           unsigned voff) {
  asm volatile("global_load_async_to_lds_b128 %0, %1, %2"
               :: "v"(lds_off), "v"(voff), "s"(sbase) : "memory");
}
__device__ __forceinline__ void wait_asynccnt0() {
  asm volatile("s_wait_asynccnt 0x0" ::: "memory");
}

// ---- one-time fp32 -> bf16 conversion of K and V into the workspace -------
__global__ __launch_bounds__(256, 1)
void cvt_bf16_kernel(const float* __restrict__ src0, const float* __restrict__ src1,
                     unsigned short* __restrict__ dst0, unsigned short* __restrict__ dst1,
                     int n)   // n = B_*S_*D_, divisible by 8*256
{
  const size_t i = ((size_t)blockIdx.x * 256 + threadIdx.x) * 8;
  const float* s;
  __bf16*      d;
  if (i < (size_t)n) { s = src0 + i;       d = (__bf16*)dst0 + i; }
  else               { s = src1 + (i - n); d = (__bf16*)dst1 + (i - n); }
  v4f a = *(const v4f*)s;
  v4f b = *(const v4f*)(s + 4);
  v8bf o;
  #pragma unroll
  for (int j = 0; j < 4; ++j) { o[j] = f2bf(a[j]); o[j + 4] = f2bf(b[j]); }
  *(v8bf*)d = o;
}

// ---- flash attention ------------------------------------------------------
__global__ __launch_bounds__(256, 1)
void fa_fwd_kernel(const float* __restrict__ Q, const float* __restrict__ K,
                   const float* __restrict__ V, const float* __restrict__ Mmask,
                   float* __restrict__ O,
                   const unsigned short* __restrict__ Kbf,
                   const unsigned short* __restrict__ Vbf,
                   int use_async)        // uniform; 1 = async bf16 path
{
  __shared__ __align__(64) __bf16 lds_k[2][BN][D_];   // 2 x 16 KB
  __shared__ __align__(64) __bf16 lds_v[2][BN][D_];   // 2 x 16 KB

  const int tid  = threadIdx.x;
  const int lane = tid & 31;
  const int wave = tid >> 5;
  const int b    = blockIdx.x / (S_ / BM);
  const int qt   = blockIdx.x % (S_ / BM);
  const int q0   = qt * BM + wave * 16;   // this wave's 16 query rows
  const int half = lane >> 4;             // 0: lanes 0-15, 1: lanes 16-31
  const int l16  = lane & 15;

  // ---- Q^T B-fragments, straight from global into registers --------------
  v16bf qTfrag[4];
  {
    const float* Qb = Q + ((size_t)b * S_ + q0) * D_;
    #pragma unroll
    for (int c = 0; c < 4; ++c) {
      const int d = 32 * c + 8 * half;
      #pragma unroll
      for (int q = 0; q < 16; ++q)
        qTfrag[c][q] = f2bf(Qb[(size_t)l16 * D_ + d + q + (q >= 8 ? 8 : 0)]);
    }
  }

  // ---- accumulators ------------------------------------------------------
  v8f oacc[8];                 // O tile: C layout, lane = dv, VGPR r = query
  #pragma unroll
  for (int n = 0; n < 8; ++n)
    #pragma unroll
    for (int r = 0; r < 8; ++r) oacc[n][r] = 0.0f;
  float m_run = -3.0e38f;      // running max (log2 domain), per lane
  float l_run = 0.0f;          // running sum, per lane

  // log2(e) folded into score scale and mask coefficient -> bare v_exp_f32
  const float scale2  = 0.088388347648318447f * LOG2E;   // log2e/sqrt(128)
  const float negbig2 = NEG_BIG * LOG2E;

  const float* Kbase = K + (size_t)b * S_ * D_;
  const float* Vbase = V + (size_t)b * S_ * D_;
  const unsigned short* KbfB = Kbf + (size_t)b * S_ * D_;
  const unsigned short* VbfB = Vbf + (size_t)b * S_ * D_;
  const float* mrow  = Mmask + ((size_t)b * S_ + (q0 + l16)) * S_;

  // loader mapping (fixed per thread)
  const int lr  = tid >> 2;           // key row within tile, 0..63
  const int lc0 = (tid & 3) * 32;     // d-chunk start

  // stage tile starting at key `kt` into LDS buffer `buf`
  auto stage = [&](int kt, int buf) {
    if (use_async) {
      const unsigned ldsK = (unsigned)(size_t)&lds_k[buf][lr][lc0];
      const unsigned ldsV = (unsigned)(size_t)&lds_v[buf][lr][lc0];
      const unsigned voff = (unsigned)(((kt + lr) * D_ + lc0) * 2);
      #pragma unroll
      for (int i = 0; i < 4; ++i) {
        async_b128(ldsK + 16 * i, KbfB, voff + 16 * i);
        async_b128(ldsV + 16 * i, VbfB, voff + 16 * i);
      }
      if (kt + BN < S_) {   // warm L2 one more tile ahead
        __builtin_prefetch(KbfB + (size_t)(kt + BN + lr) * D_ + lc0, 0, 1);
        __builtin_prefetch(VbfB + (size_t)(kt + BN + lr) * D_ + lc0, 0, 1);
      }
    } else {
      const float* krow = Kbase + (size_t)(kt + lr) * D_ + lc0;
      const float* vrow = Vbase + (size_t)(kt + lr) * D_ + lc0;
      #pragma unroll
      for (int i = 0; i < 32; i += 8) {
        v8bf pk, pv;
        #pragma unroll
        for (int j = 0; j < 8; ++j) { pk[j] = f2bf(krow[i + j]); pv[j] = f2bf(vrow[i + j]); }
        *(v8bf*)&lds_k[buf][lr][lc0 + i] = pk;
        *(v8bf*)&lds_v[buf][lr][lc0 + i] = pv;
      }
      if (kt + BN < S_) {
        __builtin_prefetch(krow + (size_t)BN * D_, 0, 1);
        __builtin_prefetch(vrow + (size_t)BN * D_, 0, 1);
      }
    }
  };

  // ---- software pipeline: preload tile 0 ---------------------------------
  stage(0, 0);
  if (use_async) wait_asynccnt0();
  __syncthreads();

  int p = 0;
  for (int k0 = 0; k0 < S_; k0 += BN, p ^= 1) {
    // issue next tile's staging into the other buffer (overlaps compute)
    if (k0 + BN < S_) stage(k0 + BN, p ^ 1);
    __builtin_prefetch(mrow + k0 + BN, 0, 1);
    __builtin_prefetch(mrow + k0 + BN + 32, 0, 1);

    // ---- S^T = K Q^T (log2-scaled) + mask  (4 sub-tiles of 16 keys) -----
    // C layout of st[t]: VGPR r -> key k0+16t + r + 8*half; lane -> query l16
    v8f st[4];
    #pragma unroll
    for (int t = 0; t < 4; ++t) {
      v8f acc;
      #pragma unroll
      for (int r = 0; r < 8; ++r) acc[r] = 0.0f;
      #pragma unroll
      for (int c = 0; c < 4; ++c) {
        v8bf lo = *(const v8bf*)&lds_k[p][16 * t + l16][32 * c + 8 * half];
        v8bf hi = *(const v8bf*)&lds_k[p][16 * t + l16][32 * c + 8 * half + 16];
        v16bf af;
        #pragma unroll
        for (int i = 0; i < 8; ++i) { af[i] = lo[i]; af[i + 8] = hi[i]; }
        acc = __builtin_amdgcn_wmma_f32_16x16x32_bf16(
                  false, af, false, qTfrag[c], (short)0, acc, false, false);
      }
      float mv[8];
      *(v4f*)&mv[0] = *(const v4f*)(mrow + k0 + 16 * t + 8 * half);
      *(v4f*)&mv[4] = *(const v4f*)(mrow + k0 + 16 * t + 8 * half + 4);
      #pragma unroll
      for (int r = 0; r < 8; ++r)
        acc[r] = acc[r] * scale2 + negbig2 * mv[r];
      st[t] = acc;
    }

    // ---- online softmax in log2 domain: per-lane + one cross-half shuffle
    float mx = st[0][0];
    #pragma unroll
    for (int t = 0; t < 4; ++t)
      #pragma unroll
      for (int r = 0; r < 8; ++r) mx = fmaxf(mx, st[t][r]);
    mx = fmaxf(mx, __shfl_xor(mx, 16, 32));   // combine key halves
    const float mnew  = fmaxf(m_run, mx);
    const float alpha = exp2f(m_run - mnew);
    m_run = mnew;
    float rsum = 0.0f;
    #pragma unroll
    for (int t = 0; t < 4; ++t)
      #pragma unroll
      for (int r = 0; r < 8; ++r) {
        float pe = exp2f(st[t][r] - mnew);
        st[t][r] = pe;
        rsum += pe;
      }
    rsum += __shfl_xor(rsum, 16, 32);
    l_run = l_run * alpha + rsum;

    // rescale O: broadcast alpha from lane-space (query=l16) into the
    // output C layout's register-space rows (query = r + 8*half)
    #pragma unroll
    for (int r = 0; r < 8; ++r) {
      const float ar = __shfl(alpha, r + 8 * half, 32);
      #pragma unroll
      for (int n = 0; n < 8; ++n) oacc[n][r] *= ar;
    }

    // ---- P A-fragments: pure register repack of st (no LDS, no shuffles)
    v16bf pfrag[2];
    #pragma unroll
    for (int c = 0; c < 2; ++c)
      #pragma unroll
      for (int i = 0; i < 8; ++i) {
        pfrag[c][i]     = f2bf(st[2 * c][i]);
        pfrag[c][i + 8] = f2bf(st[2 * c + 1][i]);
      }

    // ---- O += P V -------------------------------------------------------
    #pragma unroll
    for (int n = 0; n < 8; ++n) {
      #pragma unroll
      for (int c = 0; c < 2; ++c) {
        v16bf vf;
        #pragma unroll
        for (int s = 0; s < 16; ++s)
          vf[s] = lds_v[p][32 * c + 8 * half + s + (s >= 8 ? 8 : 0)][16 * n + l16];
        oacc[n] = __builtin_amdgcn_wmma_f32_16x16x32_bf16(
                      false, pfrag[c], false, vf, (short)0, oacc[n],
                      false, false);
      }
    }

    // next tile's async DMA must have landed before anyone reads it
    if (use_async) wait_asynccnt0();
    __syncthreads();
  }

  // ---- normalize by row sum and store fp32 -------------------------------
  const float inv = 1.0f / l_run;
  #pragma unroll
  for (int r = 0; r < 8; ++r) {
    const float ir = __shfl(inv, r + 8 * half, 32);
    float* op = O + ((size_t)b * S_ + (q0 + r + 8 * half)) * D_ + l16;
    #pragma unroll
    for (int n = 0; n < 8; ++n)
      op[n * 16] = oacc[n][r] * ir;
  }
}

extern "C" void kernel_launch(void* const* d_in, const int* in_sizes, int n_in,
                              void* d_out, int out_size, void* d_ws, size_t ws_size,
                              hipStream_t stream) {
  (void)in_sizes; (void)n_in; (void)out_size;
  const float* Q = (const float*)d_in[0];
  const float* K = (const float*)d_in[1];
  const float* V = (const float*)d_in[2];
  // d_in[3] = d_k scalar (== 128), baked into `scale`
  const float* M = (const float*)d_in[4];
  float* O = (float*)d_out;

  const int    n    = B_ * S_ * D_;                            // 4,194,304
  const size_t need = (size_t)2 * n * sizeof(unsigned short);  // 16 MB

  dim3 grid(B_ * (S_ / BM));   // 256 workgroups
  dim3 block(256);             // 8 waves (wave32)

  if (ws_size >= need) {
    unsigned short* Kbf = (unsigned short*)d_ws;
    unsigned short* Vbf = Kbf + n;
    dim3 cgrid((unsigned)((size_t)2 * n / 8 / 256));           // exact: 4096
    hipLaunchKernelGGL(cvt_bf16_kernel, cgrid, block, 0, stream, K, V, Kbf, Vbf, n);
    hipLaunchKernelGGL(fa_fwd_kernel, grid, block, 0, stream,
                       Q, K, V, M, O, Kbf, Vbf, 1);
  } else {
    hipLaunchKernelGGL(fa_fwd_kernel, grid, block, 0, stream,
                       Q, K, V, M, O, (const unsigned short*)nullptr,
                       (const unsigned short*)nullptr, 0);
  }
}
